// QCNN_simple_2bit_71141838291243
// MI455X (gfx1250) — compile-verified, matrix operand order checked by reading, not run on hardware
//
#include <hip/hip_runtime.h>
#include <hip/hip_bf16.h>
#include <math.h>

typedef _Float16 v16h __attribute__((ext_vector_type(16)));
typedef _Float16 v8h  __attribute__((ext_vector_type(8)));
typedef float    v8f  __attribute__((ext_vector_type(8)));
typedef int      v8i  __attribute__((ext_vector_type(8)));

#define BATCH 32
#define H1_H 222
#define H1_W 222
#define H2_H 220
#define H2_W 220
#define OC1 32
#define OC2 64

#define WS_SW    0
#define WS_QW1   64
#define WS_QW2   4096
#define WS_GSUM  24576
#define WS_H1    32768

__global__ __launch_bounds__(256) void k_scales(
    const float* __restrict__ w0, int n0, const float* __restrict__ w1, int n1,
    const float* __restrict__ w2, int n2, const float* __restrict__ w3, int n3,
    float* __restrict__ sw)
{
  const float* w; int n;
  switch (blockIdx.x) {
    case 0:  w = w0; n = n0; break;
    case 1:  w = w1; n = n1; break;
    case 2:  w = w2; n = n2; break;
    default: w = w3; n = n3; break;
  }
  float m = 0.f;
  for (int i = threadIdx.x; i < n; i += blockDim.x) m = fmaxf(m, fabsf(w[i]));
  __shared__ float red[256];
  red[threadIdx.x] = m;
  __syncthreads();
  for (int s = 128; s > 0; s >>= 1) {
    if (threadIdx.x < s) red[threadIdx.x] = fmaxf(red[threadIdx.x], red[threadIdx.x + s]);
    __syncthreads();
  }
  if (threadIdx.x == 0) sw[blockIdx.x] = red[0];
}

__global__ void k_pack_w1(const float* __restrict__ w, const float* __restrict__ sw,
                          _Float16* __restrict__ qw1)
{
  const int tid  = threadIdx.x;
  const int tile = tid >> 5, lane = tid & 31;
  const int h = lane >> 4, m = lane & 15;
  const float rs = 1.f / sw[0];
  _Float16* dst = qw1 + tid * 16;
  for (int p = 0; p < 8; ++p) {
    const int Kb = 16 * (p >> 2) + 2 * (p & 3) + 8 * h;
    for (int j = 0; j < 2; ++j) {
      const int K = Kb + j;
      float code = 0.f;
      if (K < 27) {
        const int tap = K / 3, ic = K - tap * 3;
        const int ty = tap / 3, tx = tap - ty * 3;
        const int oc = tile * 16 + m;
        code = rintf(fminf(fmaxf(w[((oc * 3 + ic) * 3 + ty) * 3 + tx] * rs, -1.f), 1.f));
      }
      dst[2 * p + j] = (_Float16)code;
    }
  }
}

__global__ void k_pack_w2(const float* __restrict__ w, const float* __restrict__ sw,
                          signed char* __restrict__ qw2)
{
  const int blk  = blockIdx.x;
  const int tile = blk / 5, c5 = blk % 5;
  const int lane = threadIdx.x;
  const int h = lane >> 4, m = lane & 15;
  const float rs = 1.f / sw[1];
  signed char* dst = qw2 + (size_t)(blk * 32 + lane) * 32;
  for (int v = 0; v < 8; ++v) {
    for (int j = 0; j < 4; ++j) {
      const int K = 64 * c5 + 32 * (v >> 2) + 16 * ((v >> 1) & 1) + 8 * h + 4 * (v & 1) + j;
      signed char code = 0;
      if (K < 288) {
        const int tap = K >> 5, ch = K & 31;
        const int ty = tap / 3, tx = tap - ty * 3;
        const int oc = tile * 16 + m;
        code = (signed char)(int)rintf(fminf(fmaxf(w[((oc * 32 + ch) * 3 + ty) * 3 + tx] * rs, -1.f), 1.f));
      }
      dst[v * 4 + j] = code;
    }
  }
}

__global__ void k_zero(int* __restrict__ p, int n)
{
  const int i = blockIdx.x * blockDim.x + threadIdx.x;
  if (i < n) p[i] = 0;
}

__global__ __launch_bounds__(256) void k_conv1(
    const float* __restrict__ x, const float* __restrict__ b1,
    const float* __restrict__ s1p, const float* __restrict__ sw,
    const _Float16* __restrict__ qw1, unsigned char* __restrict__ h1)
{
  __shared__ alignas(16) _Float16 imcol[224 * 32];
  const int bid = blockIdx.x;
  const int b = bid / H1_H;
  const int y = bid % H1_H;
  const int tid = threadIdx.x;

#pragma unroll
  for (int K = 0; K < 32; ++K) {
    if (tid < 224) {
      float v = 0.f;
      if (K < 27) {
        const int tap = K / 3, ic = K - tap * 3;
        const int ty = tap / 3, tx = tap - ty * 3;
        const int col = (tid + tx) < 224 ? (tid + tx) : 223;
        v = x[((size_t)(b * 3 + ic) * 224 + (y + ty)) * 224 + col];
      }
      imcol[tid * 32 + K] = (_Float16)v;
    }
  }
  __syncthreads();

  const int wave = tid >> 5;
  const int lane = tid & 31;
  const int h = lane >> 4, n = lane & 15;

  const float rs1 = 1.f / s1p[0];
  const float sc  = sw[0] * rs1;
  float t0[8], t1[8];
#pragma unroll
  for (int r = 0; r < 8; ++r) {
    t0[r] = b1[8 * h + r] * rs1;
    t1[r] = b1[16 + 8 * h + r] * rs1;
  }
  const v16h A0 = *(const v16h*)(qw1 + (0 * 32 + lane) * 16);
  const v16h A1 = *(const v16h*)(qw1 + (1 * 32 + lane) * 16);

  for (int t = wave; t < 14; t += 8) {
    const int pos = t * 16 + n;
    const _Float16* pb = imcol + pos * 32;
    const v8h lo = *(const v8h*)(pb + 8 * h);
    const v8h hi = *(const v8h*)(pb + 16 + 8 * h);
    v16h Bm;
#pragma unroll
    for (int i = 0; i < 8; ++i) { Bm[i] = lo[i]; Bm[8 + i] = hi[i]; }

    v8f c0 = {}, c1 = {};
    c0 = __builtin_amdgcn_wmma_f32_16x16x32_f16(false, A0, false, Bm, (short)0, c0, false, false);
    c1 = __builtin_amdgcn_wmma_f32_16x16x32_f16(false, A1, false, Bm, (short)0, c1, false, false);

    if (pos < H1_W) {
      unsigned int pka0 = 0, pka1 = 0, pkb0 = 0, pkb1 = 0;
#pragma unroll
      for (int r = 0; r < 4; ++r) {
        pka0 |= (unsigned int)rintf(fminf(fmaxf(c0[r] * sc + t0[r], 0.f), 3.f)) << (8 * r);
        pkb0 |= (unsigned int)rintf(fminf(fmaxf(c1[r] * sc + t1[r], 0.f), 3.f)) << (8 * r);
      }
#pragma unroll
      for (int r = 4; r < 8; ++r) {
        pka1 |= (unsigned int)rintf(fminf(fmaxf(c0[r] * sc + t0[r], 0.f), 3.f)) << (8 * (r - 4));
        pkb1 |= (unsigned int)rintf(fminf(fmaxf(c1[r] * sc + t1[r], 0.f), 3.f)) << (8 * (r - 4));
      }
      unsigned char* base = h1 + ((size_t)(b * H1_H + y) * H1_W + pos) * 32;
      uint2 va; va.x = pka0; va.y = pka1;
      uint2 vb; vb.x = pkb0; vb.y = pkb1;
      *(uint2*)(base + 8 * h)      = va;
      *(uint2*)(base + 16 + 8 * h) = vb;
    }
  }
}

__global__ __launch_bounds__(256) void k_conv2(
    const unsigned char* __restrict__ h1, const signed char* __restrict__ qw2,
    const float* __restrict__ b2, const float* __restrict__ s1p,
    const float* __restrict__ s2p, const float* __restrict__ sw,
    int* __restrict__ gsum)
{
  __shared__ alignas(32) unsigned char h1s[3 * H1_W * 32];
  __shared__ alignas(32) signed char   qw2s[20 * 32 * 32];
  __shared__ int pool[OC2];

  const int bid = blockIdx.x;
  const int b = bid / H2_H;
  const int y = bid % H2_H;
  const int tid = threadIdx.x;

  {
    const uint4* src = (const uint4*)(h1 + (size_t)(b * H1_H + y) * H1_W * 32);
    uint4* dst = (uint4*)h1s;
    for (int i = tid; i < (3 * H1_W * 32) / 16; i += 256) dst[i] = src[i];
    const uint4* wsrc = (const uint4*)qw2;
    uint4* wdst = (uint4*)qw2s;
    for (int i = tid; i < (20 * 32 * 32) / 16; i += 256) wdst[i] = wsrc[i];
    if (tid < OC2) pool[tid] = 0;
  }
  __syncthreads();

  const int wave = tid >> 5;
  const int lane = tid & 31;
  const int h = lane >> 4, n = lane & 15;

  if (wave < 7) {
    const float rs2 = 1.f / s2p[0];
    const float sc2 = s1p[0] * sw[1] * rs2;
    float toff[4][8];
#pragma unroll
    for (int tt = 0; tt < 4; ++tt)
#pragma unroll
      for (int r = 0; r < 8; ++r)
        toff[tt][r] = b2[tt * 16 + 8 * h + r] * rs2;

    int  pxs[2]; bool vals[2];
#pragma unroll
    for (int s = 0; s < 2; ++s) {
      const int pos = wave * 32 + s * 16 + n;
      vals[s] = pos < H2_W;
      pxs[s]  = vals[s] ? pos : 0;
    }

    v8i acc[2][4] = {};
#pragma unroll
    for (int c5 = 0; c5 < 5; ++c5) {
      v8i Bm[2];
#pragma unroll
      for (int s = 0; s < 2; ++s) {
#pragma unroll
        for (int q = 0; q < 4; ++q) {
          const int K0 = 64 * c5 + 16 * q;
          unsigned long long d = 0;
          if (K0 < 288) {
            const int tap = K0 >> 5;
            const int ty = tap / 3, tx = tap - ty * 3;
            const int chb = K0 & 31;
            d = *(const unsigned long long*)(h1s + ((ty * H1_W) + (pxs[s] + tx)) * 32 + chb + 8 * h);
          }
          Bm[s][2 * q]     = (int)(unsigned int)(d & 0xffffffffull);
          Bm[s][2 * q + 1] = (int)(unsigned int)(d >> 32);
        }
      }
#pragma unroll
      for (int tt = 0; tt < 4; ++tt) {
        const v8i A = *(const v8i*)(qw2s + (size_t)((tt * 5 + c5) * 32 + lane) * 32);
        acc[0][tt] = __builtin_amdgcn_wmma_i32_16x16x64_iu8(true, A, false, Bm[0], acc[0][tt], false, false);
        acc[1][tt] = __builtin_amdgcn_wmma_i32_16x16x64_iu8(true, A, false, Bm[1], acc[1][tt], false, false);
      }
    }

#pragma unroll
    for (int s = 0; s < 2; ++s) {
#pragma unroll
      for (int tt = 0; tt < 4; ++tt) {
        int p0 = 0, p1 = 0;
#pragma unroll
        for (int r = 0; r < 8; ++r) {
          int q = (int)rintf(fminf(fmaxf((float)acc[s][tt][r] * sc2 + toff[tt][r], 0.f), 3.f));
          if (!vals[s]) q = 0;
          if (r < 4) p0 |= q << (8 * r);
          else       p1 |= q << (8 * (r - 4));
        }
#pragma unroll
        for (int off = 1; off < 16; off <<= 1) {
          p0 += __shfl_xor(p0, off, 32);
          p1 += __shfl_xor(p1, off, 32);
        }
        if (n == 0) {
#pragma unroll
          for (int r = 0; r < 4; ++r) {
            atomicAdd(&pool[tt * 16 + 8 * h + r],     (p0 >> (8 * r)) & 0xFF);
            atomicAdd(&pool[tt * 16 + 8 * h + 4 + r], (p1 >> (8 * r)) & 0xFF);
          }
        }
      }
    }
  }
  __syncthreads();
  if (tid < OC2) atomicAdd(&gsum[b * OC2 + tid], pool[tid]);
}

__global__ __launch_bounds__(256) void k_head(
    const int* __restrict__ gsum,
    const float* __restrict__ fc1_w, const float* __restrict__ fc1_b,
    const float* __restrict__ fc2_w, const float* __restrict__ fc2_b,
    const float* __restrict__ s2p, const float* __restrict__ s3p,
    const float* __restrict__ sw, float* __restrict__ out)
{
  __shared__ float pooled[BATCH * OC2];
  __shared__ float a3[BATCH * 128];
  __shared__ float logits[BATCH * 10];
  const int tid = threadIdx.x;
  const float s2 = s2p[0], s3 = s3p[0];
  const float rs3 = 1.f / s3;
  const float sf1 = sw[2], sf2 = sw[3];
  const float rsf1 = 1.f / sf1, rsf2 = 1.f / sf2;
  const float inv = 1.f / ((float)H2_H * (float)H2_W);

  for (int i = tid; i < BATCH * OC2; i += 256) pooled[i] = s2 * (float)gsum[i] * inv;
  __syncthreads();
  for (int i = tid; i < BATCH * 128; i += 256) {
    const int bb = i >> 7, j = i & 127;
    float acc = fc1_b[j];
    for (int c = 0; c < 64; ++c) {
      const float wq = rintf(fminf(fmaxf(fc1_w[j * 64 + c] * rsf1, -1.f), 1.f)) * sf1;
      acc += pooled[bb * 64 + c] * wq;
    }
    a3[i] = rintf(fminf(fmaxf(acc * rs3, 0.f), 3.f)) * s3;
  }
  __syncthreads();
  for (int i = tid; i < BATCH * 10; i += 256) {
    const int bb = i / 10, k = i - bb * 10;
    float acc = fc2_b[k];
    for (int c = 0; c < 128; ++c) {
      const float wq = rintf(fminf(fmaxf(fc2_w[k * 128 + c] * rsf2, -1.f), 1.f)) * sf2;
      acc += a3[bb * 128 + c] * wq;
    }
    logits[i] = acc;
  }
  __syncthreads();
  if (tid < BATCH) {
    float m = -1e30f;
    for (int k = 0; k < 10; ++k) m = fmaxf(m, logits[tid * 10 + k]);
    float se = 0.f;
    for (int k = 0; k < 10; ++k) se += expf(logits[tid * 10 + k] - m);
    const float lse = logf(se);
    for (int k = 0; k < 10; ++k) out[tid * 10 + k] = logits[tid * 10 + k] - m - lse;
  }
}

extern "C" void kernel_launch(void* const* d_in, const int* in_sizes, int n_in,
                              void* d_out, int out_size, void* d_ws, size_t ws_size,
                              hipStream_t stream)
{
  const float* x   = (const float*)d_in[0];
  const float* w1  = (const float*)d_in[1];
  const float* b1  = (const float*)d_in[2];
  const float* w2  = (const float*)d_in[3];
  const float* b2  = (const float*)d_in[4];
  const float* fw1 = (const float*)d_in[5];
  const float* fb1 = (const float*)d_in[6];
  const float* fw2 = (const float*)d_in[7];
  const float* fb2 = (const float*)d_in[8];
  const float* s1  = (const float*)d_in[9];
  const float* s2  = (const float*)d_in[10];
  const float* s3  = (const float*)d_in[11];

  char* ws = (char*)d_ws;
  float*         sw   = (float*)(ws + WS_SW);
  _Float16*      qw1  = (_Float16*)(ws + WS_QW1);
  signed char*   qw2  = (signed char*)(ws + WS_QW2);
  int*           gsum = (int*)(ws + WS_GSUM);
  unsigned char* h1   = (unsigned char*)(ws + WS_H1);

  k_scales<<<4, 256, 0, stream>>>(w1, 32 * 3 * 9, w2, 64 * 32 * 9,
                                  fw1, 128 * 64, fw2, 10 * 128, sw);
  k_pack_w1<<<1, 64, 0, stream>>>(w1, sw, qw1);
  k_pack_w2<<<20, 32, 0, stream>>>(w2, sw, qw2);
  k_zero<<<8, 256, 0, stream>>>(gsum, BATCH * OC2);
  k_conv1<<<BATCH * H1_H, 256, 0, stream>>>(x, b1, s1, sw, qw1, h1);
  k_conv2<<<BATCH * H2_H, 256, 0, stream>>>(h1, qw2, b2, s1, s2, sw, gsum);
  k_head<<<1, 256, 0, stream>>>(gsum, fw1, fb1, fw2, fb2, s2, s3, sw, (float*)d_out);
}
